// Self_Attn_18356690223645
// MI455X (gfx1250) — matrix operand order repaired and checked by reading, not compiled, not run on hardware
//
#include <hip/hip_runtime.h>

#define DEVI __device__ __forceinline__

typedef __attribute__((ext_vector_type(16))) __bf16 bf16x16;
typedef __attribute__((ext_vector_type(4)))  __bf16 bf16x4;
typedef __attribute__((ext_vector_type(8)))  float  f32x8;
typedef __attribute__((ext_vector_type(4)))  int    i32x4;

// ---- problem constants ----
#define BB   2
#define CC   256
#define SS   32
#define NN   1024          // S*S
#define C8   32
#define OUT_ELEMS (BB*CC*NN)   // 524288 floats, then attn (BB*NN*NN)

// ---- LDS strides (bank-conflict padding, 64 banks x 4B) ----
#define XL_STRIDE   144    // bf16 elems per row (128 data + 16 pad) -> 8-word shift/row
#define WA_STRIDE   272    // bf16 elems per row (256 data + 16 pad) -> 8-word shift/row
#define AT_STRIDE   1028   // f32 elems per row  (1024 data + 4 pad)

// ---- CDNA5 async global->LDS copies (ASYNCcnt), guarded by __has_builtin ----
#if defined(__has_builtin)
#if __has_builtin(__builtin_amdgcn_global_load_async_to_lds_b128) && \
    __has_builtin(__builtin_amdgcn_s_wait_asynccnt)
#define USE_ASYNC_LDS 1
#endif
#endif

#ifdef USE_ASYNC_LDS
typedef __attribute__((address_space(1))) i32x4 gi32x4;   // global int4
typedef __attribute__((address_space(3))) i32x4 li32x4;   // LDS int4
#endif

DEVI f32x8 wmma_bf16(bf16x16 a, bf16x16 b, f32x8 c) {
  return __builtin_amdgcn_wmma_f32_16x16x32_bf16(false, a, false, b, (short)0, c,
                                                 false, false);
}

DEVI bf16x16 ld32B(const __bf16* p) {            // 32 contiguous bytes
  bf16x16 r;
  ((uint4*)&r)[0] = ((const uint4*)p)[0];
  ((uint4*)&r)[1] = ((const uint4*)p)[1];
  return r;
}
DEVI bf16x16 ld16Bx2(const __bf16* p0, const __bf16* p1) {  // two 16B chunks
  bf16x16 r;
  ((uint4*)&r)[0] = *(const uint4*)p0;
  ((uint4*)&r)[1] = *(const uint4*)p1;
  return r;
}
DEVI bf16x16 cvt16(const float* p0, const float* p1) {      // 8+8 f32 -> bf16
  float4 a0 = ((const float4*)p0)[0], a1 = ((const float4*)p0)[1];
  float4 b0 = ((const float4*)p1)[0], b1 = ((const float4*)p1)[1];
  bf16x16 r;
  r[0]=(__bf16)a0.x; r[1]=(__bf16)a0.y; r[2]=(__bf16)a0.z; r[3]=(__bf16)a0.w;
  r[4]=(__bf16)a1.x; r[5]=(__bf16)a1.y; r[6]=(__bf16)a1.z; r[7]=(__bf16)a1.w;
  r[8]=(__bf16)b0.x; r[9]=(__bf16)b0.y; r[10]=(__bf16)b0.z; r[11]=(__bf16)b0.w;
  r[12]=(__bf16)b1.x; r[13]=(__bf16)b1.y; r[14]=(__bf16)b1.z; r[15]=(__bf16)b1.w;
  return r;
}
DEVI void cvt_store4(__bf16* d, float4 v) {
  bf16x4 t; t[0]=(__bf16)v.x; t[1]=(__bf16)v.y; t[2]=(__bf16)v.z; t[3]=(__bf16)v.w;
  *(bf16x4*)d = t;
}

// ---------------------------------------------------------------------------
// Kernel 1: cast x -> bf16 and build spatial-transposed copy XT (ws)
// ---------------------------------------------------------------------------
__global__ void prep_kernel(const float* __restrict__ x,
                            __bf16* __restrict__ Xbf, __bf16* __restrict__ XTbf) {
  int idx = blockIdx.x * 256 + threadIdx.x;       // < B*C*N
  int p  = idx & (NN - 1);
  int bc = idx >> 10;
  int u = p >> 5, v = p & 31;
  int pT = v * SS + u;
  Xbf[idx]                       = (__bf16)x[idx];
  XTbf[(size_t)bc * NN + p]      = (__bf16)x[(size_t)bc * NN + pT];
}

// ---------------------------------------------------------------------------
// Kernel 2: h = Wh @ X + bh   (M=256, N=1024, K=256), one 16x16 tile per wave
// ---------------------------------------------------------------------------
__global__ void h_kernel(const float* __restrict__ Wh, const float* __restrict__ bh,
                         const __bf16* __restrict__ Xbf, float* __restrict__ hout) {
  int tid  = threadIdx.x;
  int wave = tid >> 5, lane = tid & 31;
  int tile = blockIdx.x * 8 + wave;               // 0..2047
  int b  = tile >> 10;
  int rm = tile & 1023;
  int ct = rm >> 6;                               // c-tile 0..15
  int pt = rm & 63;                               // p-tile 0..63
  int m = lane & 15, hh = lane >> 4;

  const __bf16* Xb = Xbf + (size_t)b * CC * NN;
  f32x8 acc = {};
#pragma unroll
  for (int cb = 0; cb < 8; ++cb) {
    int c0 = cb * 32;
    const float* ar = Wh + (size_t)(ct * 16 + m) * CC + c0 + hh * 8;
    bf16x16 A = cvt16(ar, ar + 16);
    bf16x16 Bv;
    for (int s = 0; s < 16; ++s) Bv[s] = Xb[(size_t)(c0 + hh * 8 + s + (s & 8)) * NN + pt * 16 + m];
    acc = wmma_bf16(A, Bv, acc);
  }
#pragma unroll
  for (int e = 0; e < 8; ++e) {
    int c = ct * 16 + hh * 8 + e;
    hout[((size_t)b * CC + c) * NN + pt * 16 + m] = acc[e] + bh[c];
  }
}

// ---------------------------------------------------------------------------
// Kernel 3 (main): fused f/g GEMMs + k-contraction + softmax
// One workgroup per (b, i-block of 16). 8 waves. Dynamic LDS ~276KB.
// ---------------------------------------------------------------------------
__global__ void attn_kernel(const float* __restrict__ Wf, const float* __restrict__ bfb,
                            const float* __restrict__ Wg, const float* __restrict__ bgb,
                            const __bf16* __restrict__ Xbf, const __bf16* __restrict__ XTbf,
                            float* __restrict__ attn_out) {
  extern __shared__ __align__(16) char smem[];
  __bf16* XL  = (__bf16*)smem;                          // 256 x XL_STRIDE
  __bf16* XTL = XL  + 256 * XL_STRIDE;
  __bf16* WfL = XTL + 256 * XL_STRIDE;                  // 64 x WA_STRIDE (4 k's)
  __bf16* WgL = WfL + 64 * WA_STRIDE;
  float*  AT  = (float*)(WgL + 64 * WA_STRIDE);         // 16 x AT_STRIDE

  const int tid  = threadIdx.x;
  const int wave = tid >> 5, lane = tid & 31;
  const int b  = blockIdx.x & 1;                        // b fastest -> L2 reuse of Wf/Wg
  const int i0 = (blockIdx.x >> 1) * 16;
  const int m  = lane & 15;
  const int hh = lane >> 4;

  const __bf16* Xb  = Xbf  + (size_t)b * CC * NN;
  const __bf16* XTb = XTbf + (size_t)b * CC * NN;

  for (int jc = 0; jc < 8; ++jc) {                      // 128-column j-chunks
    __syncthreads();                                    // prior chunk done with XL/XTL
    {
      // stage X/XT chunk: thread t copies row t (128 bf16 = 16 x b128)
#ifdef USE_ASYNC_LDS
      gi32x4* s0 = (gi32x4*)(Xb  + (size_t)tid * NN + jc * 128);
      gi32x4* s1 = (gi32x4*)(XTb + (size_t)tid * NN + jc * 128);
      li32x4* d0 = (li32x4*)(XL  + tid * XL_STRIDE);
      li32x4* d1 = (li32x4*)(XTL + tid * XL_STRIDE);
#pragma unroll
      for (int q = 0; q < 16; ++q) {
        __builtin_amdgcn_global_load_async_to_lds_b128(s0 + q, d0 + q, 0, 0);
        __builtin_amdgcn_global_load_async_to_lds_b128(s1 + q, d1 + q, 0, 0);
      }
#else
      const uint4* s0 = (const uint4*)(Xb  + (size_t)tid * NN + jc * 128);
      const uint4* s1 = (const uint4*)(XTb + (size_t)tid * NN + jc * 128);
      uint4* d0 = (uint4*)(XL  + tid * XL_STRIDE);
      uint4* d1 = (uint4*)(XTL + tid * XL_STRIDE);
#pragma unroll
      for (int q = 0; q < 16; ++q) { d0[q] = s0[q]; d1[q] = s1[q]; }
#endif
    }

    f32x8 acc = {};                                     // attn tile accumulator (over k)
    for (int kb = 0; kb < 8; ++kb) {                    // 4 k's per stage
      __syncthreads();                                  // WfL/WgL free (also X ready @kb=0)
      {
        int row = tid >> 2;                             // 0..63 : (kk,m)
        int cb0 = (tid & 3) * 64;
        int kk = row >> 4, mm = row & 15;
        int grow = ((kb * 4 + kk) * NN + i0 + mm) * CC;
        const float4* sf = (const float4*)(Wf + grow + cb0);
        const float4* sg = (const float4*)(Wg + grow + cb0);
        __bf16* df = WfL + row * WA_STRIDE + cb0;
        __bf16* dg = WgL + row * WA_STRIDE + cb0;
#pragma unroll
        for (int q = 0; q < 16; ++q) {
          cvt_store4(df + q * 4, sf[q]);
          cvt_store4(dg + q * 4, sg[q]);
        }
        if (kb < 7) {                                   // global_prefetch_b8 next panel
          __builtin_prefetch(Wf + grow + 4 * NN * CC + cb0, 0, 0);
          __builtin_prefetch(Wg + grow + 4 * NN * CC + cb0, 0, 0);
        }
      }
#ifdef USE_ASYNC_LDS
      if (kb == 0) __builtin_amdgcn_s_wait_asynccnt(0); // X/XT DMA drained (overlapped W stage)
#endif
      __syncthreads();

#pragma unroll
      for (int kk = 0; kk < 4; ++kk) {
        const int k = kb * 4 + kk;
        f32x8 fAcc = {}, gAcc = {};
#pragma unroll
        for (int cb = 0; cb < 8; ++cb) {
          int c0 = cb * 32;
          const __bf16* af = WfL + (kk * 16 + m) * WA_STRIDE + c0 + hh * 8;
          const __bf16* ag = WgL + (kk * 16 + m) * WA_STRIDE + c0 + hh * 8;
          bf16x16 Af = ld16Bx2(af, af + 16);
          bf16x16 Ag = ld16Bx2(ag, ag + 16);
          bf16x16 Bf, Bg;
          for (int s = 0; s < 16; ++s) {
            Bf[s] = XTL[(c0 + hh * 8 + s + (s & 8)) * XL_STRIDE + wave * 16 + m];
            Bg[s] = XL [(c0 + hh * 8 + s + (s & 8)) * XL_STRIDE + wave * 16 + m];
          }
          fAcc = wmma_bf16(Af, Bf, fAcc);
          gAcc = wmma_bf16(Ag, Bg, gAcc);
        }
        const float* bfp = bfb + k * NN + i0 + hh * 8;  // bias rows m = 8*hh+e
        const float* bgp = bgb + k * NN + i0 + hh * 8;
#pragma unroll
        for (int e = 0; e < 8; ++e)
          acc[e] += (fAcc[e] + bfp[e]) * (gAcc[e] + bgp[e]);
      }
    }
    // scatter tile into attn row buffer: D[mrow=8*hh+e, n=m] @ col jc*128+wave*16+m
#pragma unroll
    for (int e = 0; e < 8; ++e)
      AT[(hh * 8 + e) * AT_STRIDE + jc * 128 + wave * 16 + m] = acc[e];
  }
  __syncthreads();

  // softmax: each wave handles rows {wave, wave+8}
  for (int r = wave; r < 16; r += 8) {
    const float* row = AT + r * AT_STRIDE;
    float mx = -1e30f;
#pragma unroll
    for (int t = 0; t < 32; ++t) mx = fmaxf(mx, row[lane + t * 32]);
    for (int off = 16; off > 0; off >>= 1) mx = fmaxf(mx, __shfl_xor(mx, off, 32));
    float ev[32];
    float sum = 0.f;
#pragma unroll
    for (int t = 0; t < 32; ++t) { float e_ = __expf(row[lane + t * 32] - mx); ev[t] = e_; sum += e_; }
    for (int off = 16; off > 0; off >>= 1) sum += __shfl_xor(sum, off, 32);
    float inv = 1.0f / sum;
    float* orow = attn_out + ((size_t)b * NN + i0 + r) * NN;
#pragma unroll
    for (int t = 0; t < 32; ++t) orow[lane + t * 32] = ev[t] * inv;
  }
}

// ---------------------------------------------------------------------------
// Kernel 4: out = gamma * (h @ attn^T) + x
// grid: (b, i-tile, c-half); each wave one 16(c) x 16(i) tile, K = j = 1024
// ---------------------------------------------------------------------------
__global__ void out_kernel(const float* __restrict__ hws, const float* __restrict__ attn,
                           const float* __restrict__ x, const float* __restrict__ gamma,
                           float* __restrict__ out) {
  __shared__ __align__(32) __bf16 BL[32 * 16];          // attn tile, n-major, stride 32 bf16
  int tid  = threadIdx.x;
  int wave = tid >> 5, lane = tid & 31;
  int bid = blockIdx.x;
  int b  = bid & 1;
  int it = (bid >> 1) & 63;
  int ch = bid >> 7;                                    // c-half 0/1
  int ct = ch * 8 + wave;
  int m = lane & 15, hh = lane >> 4;

  const float* hb = hws  + (size_t)b * CC * NN;
  const float* ab = attn + (size_t)b * NN * NN;
  f32x8 acc = {};
  for (int jb = 0; jb < 32; ++jb) {                     // K blocks of 32
    __syncthreads();
    for (int idx = tid; idx < 512; idx += 256) {
      int k = idx & 31, n = idx >> 5;
      BL[n * 32 + k] = (__bf16)ab[(size_t)(it * 16 + n) * NN + jb * 32 + k];
    }
    __syncthreads();
    bf16x16 Bv = ld16Bx2(&BL[m * 32 + hh * 8], &BL[m * 32 + hh * 8 + 16]);
    const float* ar = hb + (size_t)(ct * 16 + m) * NN + jb * 32 + hh * 8;
    bf16x16 A = cvt16(ar, ar + 16);
    acc = wmma_bf16(A, Bv, acc);
  }
  float g = gamma[0];
#pragma unroll
  for (int e = 0; e < 8; ++e) {
    int c = ct * 16 + hh * 8 + e;
    size_t o = ((size_t)b * CC + c) * NN + it * 16 + m;
    out[o] = g * acc[e] + x[o];
  }
}

// ---------------------------------------------------------------------------
extern "C" void kernel_launch(void* const* d_in, const int* in_sizes, int n_in,
                              void* d_out, int out_size, void* d_ws, size_t ws_size,
                              hipStream_t stream) {
  const float* x     = (const float*)d_in[0];
  const float* Wf    = (const float*)d_in[1];
  const float* bf_   = (const float*)d_in[2];
  const float* Wg    = (const float*)d_in[3];
  const float* bg_   = (const float*)d_in[4];
  const float* Wh    = (const float*)d_in[5];
  const float* bh    = (const float*)d_in[6];
  const float* gamma = (const float*)d_in[7];

  float* out      = (float*)d_out;                 // (2,256,32,32)
  float* attn_out = out + OUT_ELEMS;               // (2,1,1024,1024)

  __bf16* Xbf  = (__bf16*)d_ws;                    // 1 MB
  __bf16* XTbf = Xbf + (size_t)BB * CC * NN;       // 1 MB
  float*  hws  = (float*)((char*)d_ws + (size_t)2 * BB * CC * NN * sizeof(__bf16)); // 2 MB

  // prep: cast + spatial transpose
  prep_kernel<<<(BB * CC * NN) / 256, 256, 0, stream>>>(x, Xbf, XTbf);
  // h = Wh X + bh
  h_kernel<<<256, 256, 0, stream>>>(Wh, bh, Xbf, hws);
  // fused attention logits + softmax
  const int smemB = 2 * (256 * XL_STRIDE * 2) + 2 * (64 * WA_STRIDE * 2)
                  + 16 * AT_STRIDE * 4;            // = 282,880 bytes (< 320KB WGP LDS)
  (void)hipFuncSetAttribute((const void*)attn_kernel,
                            hipFuncAttributeMaxDynamicSharedMemorySize, smemB);
  attn_kernel<<<BB * (NN / 16), 256, smemB, stream>>>(Wf, bf_, Wg, bg_, Xbf, XTbf, attn_out);
  // out = gamma * h @ attn^T + x
  out_kernel<<<BB * 64 * 2, 256, 0, stream>>>(hws, attn_out, x, gamma, out);
}
